// MyGenerator_62337155334336
// MI455X (gfx1250) — compile-verified, matrix operand order checked by reading, not run on hardware
//
#include <hip/hip_runtime.h>
#include <hip/hip_bf16.h>

typedef __attribute__((ext_vector_type(16))) _Float16 v16h;
typedef __attribute__((ext_vector_type(8)))  _Float16 v8h;
typedef __attribute__((ext_vector_type(8)))  float    v8f;
typedef __attribute__((ext_vector_type(4)))  unsigned int u32x4;
typedef __attribute__((ext_vector_type(8)))  int i32x8;
typedef __attribute__((ext_vector_type(4)))  int i32x4;

#define TPB 256
#define GRID1(n) dim3((unsigned)(((n) + TPB - 1) / TPB))

#if defined(__HIP_DEVICE_COMPILE__) && __has_builtin(__builtin_amdgcn_tensor_load_to_lds)
  #define USE_TDM 1
  #if __has_include(<hip/amd_detail/amd_gfx1250_TDM.h>)
    #define TDM_ARITY 6
  #else
    #define TDM_ARITY 5
  #endif
#else
  #define USE_TDM 0
#endif

// ---------------------------------------------------------------------------
// Reflect-pad(3) 7x7 conv, stride 1, square HxH. act: 0=none, 1=tanh
// ---------------------------------------------------------------------------
__global__ void k_refpad_conv7(const float* __restrict__ x, const float* __restrict__ w,
                               const float* __restrict__ bias, float* __restrict__ y,
                               int B, int IC, int OC, int H, int act)
{
    long long idx = (long long)blockIdx.x * blockDim.x + threadIdx.x;
    long long total = (long long)B * OC * H * H;
    if (idx >= total) return;
    int ox = (int)(idx % H); long long t = idx / H;
    int oy = (int)(t % H); t /= H;
    int oc = (int)(t % OC); int b = (int)(t / OC);
    float acc = bias[oc];
    for (int ic = 0; ic < IC; ++ic) {
        const float* xp = x + ((size_t)(b * IC + ic)) * H * H;
        const float* wp = w + ((size_t)(oc * IC + ic)) * 49;
        for (int kh = 0; kh < 7; ++kh) {
            int iy = oy - 3 + kh;
            if (iy < 0) iy = -iy;
            if (iy >= H) iy = 2 * H - 2 - iy;
            for (int kw = 0; kw < 7; ++kw) {
                int ix = ox - 3 + kw;
                if (ix < 0) ix = -ix;
                if (ix >= H) ix = 2 * H - 2 - ix;
                acc += xp[iy * H + ix] * wp[kh * 7 + kw];
            }
        }
    }
    if (act == 1) acc = tanhf(acc);
    y[idx] = acc;
}

// ---------------------------------------------------------------------------
// Generic direct conv (used for 3x3 s2 p1 and 1x1 s1 p0)
// ---------------------------------------------------------------------------
__global__ void k_conv2d(const float* __restrict__ x, const float* __restrict__ w,
                         const float* __restrict__ bias, float* __restrict__ y,
                         int B, int IC, int IH, int IW, int OC, int OH, int OW,
                         int KH, int KW, int stride, int pad)
{
    long long idx = (long long)blockIdx.x * blockDim.x + threadIdx.x;
    long long total = (long long)B * OC * OH * OW;
    if (idx >= total) return;
    int ox = (int)(idx % OW); long long t = idx / OW;
    int oy = (int)(t % OH); t /= OH;
    int oc = (int)(t % OC); int b = (int)(t / OC);
    float acc = bias[oc];
    for (int ic = 0; ic < IC; ++ic) {
        const float* xp = x + ((size_t)(b * IC + ic)) * IH * IW;
        const float* wp = w + ((size_t)(oc * IC + ic)) * KH * KW;
        for (int kh = 0; kh < KH; ++kh) {
            int iy = oy * stride - pad + kh;
            if (iy < 0 || iy >= IH) continue;
            for (int kw = 0; kw < KW; ++kw) {
                int ix = ox * stride - pad + kw;
                if (ix < 0 || ix >= IW) continue;
                acc += xp[iy * IW + ix] * wp[kh * KW + kw];
            }
        }
    }
    y[idx] = acc;
}

// ---------------------------------------------------------------------------
// lax.conv_general_dilated with lhs_dilation=2, padding=((1,2),(1,2)), 3x3 OIHW
// ---------------------------------------------------------------------------
__global__ void k_tconv3(const float* __restrict__ x, const float* __restrict__ w,
                         const float* __restrict__ bias, float* __restrict__ y,
                         int B, int IC, int IH, int IW, int OC)
{
    int OH = 2 * IH, OW = 2 * IW;
    long long idx = (long long)blockIdx.x * blockDim.x + threadIdx.x;
    long long total = (long long)B * OC * OH * OW;
    if (idx >= total) return;
    int ox = (int)(idx % OW); long long t = idx / OW;
    int oy = (int)(t % OH); t /= OH;
    int oc = (int)(t % OC); int b = (int)(t / OC);
    float acc = bias[oc];
    for (int kh = 0; kh < 3; ++kh) {
        int ty = oy + kh - 1;
        if (ty < 0 || (ty & 1)) continue;
        int iy = ty >> 1;
        if (iy >= IH) continue;
        for (int kw = 0; kw < 3; ++kw) {
            int tx = ox + kw - 1;
            if (tx < 0 || (tx & 1)) continue;
            int ix = tx >> 1;
            if (ix >= IW) continue;
            const float* xp = x + (((size_t)b * IC) * IH + iy) * IW + ix;
            const float* wp = w + ((size_t)oc * IC) * 9 + kh * 3 + kw;
            for (int ic = 0; ic < IC; ++ic)
                acc += xp[(size_t)ic * IH * IW] * wp[(size_t)ic * 9];
        }
    }
    y[idx] = acc;
}

// ---------------------------------------------------------------------------
// In-place instance norm (+ leaky relu 0.2). grid = B*C
// ---------------------------------------------------------------------------
__global__ void k_inorm(float* __restrict__ y, int HW, int leaky)
{
    __shared__ float s1[TPB], s2[TPB];
    float* p = y + (size_t)blockIdx.x * HW;
    int tid = threadIdx.x;
    float a = 0.f, b2 = 0.f;
    for (int i = tid; i < HW; i += TPB) { float v = p[i]; a += v; b2 += v * v; }
    s1[tid] = a; s2[tid] = b2; __syncthreads();
    for (int s = TPB / 2; s > 0; s >>= 1) {
        if (tid < s) { s1[tid] += s1[tid + s]; s2[tid] += s2[tid + s]; }
        __syncthreads();
    }
    float mean = s1[0] / (float)HW;
    float var  = s2[0] / (float)HW - mean * mean;
    float rstd = rsqrtf(var + 1e-5f);
    for (int i = tid; i < HW; i += TPB) {
        float v = (p[i] - mean) * rstd;
        if (leaky && v < 0.f) v *= 0.2f;
        p[i] = v;
    }
}

__global__ void k_add(float* __restrict__ dst, const float* __restrict__ src, long long n)
{
    long long i = (long long)blockIdx.x * blockDim.x + threadIdx.x;
    if (i < n) dst[i] += src[i];
}

__global__ void k_zero(float* __restrict__ p, long long n)
{
    long long i = (long long)blockIdx.x * blockDim.x + threadIdx.x;
    if (i < n) p[i] = 0.f;
}

// ---------------------------------------------------------------------------
// jax.image.resize 'linear', antialias=False
// ---------------------------------------------------------------------------
__global__ void k_resize_bilinear(const float* __restrict__ x, float* __restrict__ y,
                                  int B, int C, int IH, int IW, int OH, int OW)
{
    long long idx = (long long)blockIdx.x * blockDim.x + threadIdx.x;
    long long total = (long long)B * C * OH * OW;
    if (idx >= total) return;
    int ox = (int)(idx % OW); long long t = idx / OW;
    int oy = (int)(t % OH); t /= OH;
    int bc = (int)t;
    float cy = (oy + 0.5f) * ((float)IH / (float)OH) - 0.5f;
    float cx = (ox + 0.5f) * ((float)IW / (float)OW) - 0.5f;
    int y0 = (int)floorf(cy), x0 = (int)floorf(cx);
    float fy = cy - (float)y0, fx = cx - (float)x0;
    int y1 = y0 + 1, x1 = x0 + 1;
    y0 = min(max(y0, 0), IH - 1); y1 = min(max(y1, 0), IH - 1);
    x0 = min(max(x0, 0), IW - 1); x1 = min(max(x1, 0), IW - 1);
    const float* p = x + (size_t)bc * IH * IW;
    float v00 = p[y0 * IW + x0], v01 = p[y0 * IW + x1];
    float v10 = p[y1 * IW + x0], v11 = p[y1 * IW + x1];
    y[idx] = (v00 * (1.f - fx) + v01 * fx) * (1.f - fy) + (v10 * (1.f - fx) + v11 * fx) * fy;
}

// ---------------------------------------------------------------------------
// Quarter-res mask counts for regions 1..5. grid (5,2)
// ---------------------------------------------------------------------------
__global__ void k_mask_count(const float* __restrict__ mask, float* __restrict__ cnt)
{
    int r = blockIdx.x + 1, b = blockIdx.y, tid = threadIdx.x;
    __shared__ float s[TPB];
    float a = 0.f;
    for (int p = tid; p < 4096; p += TPB) {
        int yq = p >> 6, xq = p & 63;
        a += mask[(((size_t)b * 6 + r) * 256 + yq * 4) * 256 + xq * 4];
    }
    s[tid] = a; __syncthreads();
    for (int st = TPB / 2; st > 0; st >>= 1) {
        if (tid < st) s[tid] += s[tid + st];
        __syncthreads();
    }
    if (tid == 0) cnt[b * 5 + r - 1] = s[0];
}

// ---------------------------------------------------------------------------
// Per-region masked per-channel mean. grid (512,2)
// ---------------------------------------------------------------------------
__global__ void k_region_mean(const float* __restrict__ feat, const float* __restrict__ mask,
                              const float* __restrict__ cnt, float* __restrict__ mean, int region)
{
    int ch = blockIdx.x, b = blockIdx.y, tid = threadIdx.x;
    __shared__ float s[TPB];
    const float* f = feat + ((size_t)b * 512 + ch) * 4096;
    float a = 0.f;
    for (int p = tid; p < 4096; p += TPB) {
        int yq = p >> 6, xq = p & 63;
        float m = mask[(((size_t)b * 6 + region) * 256 + yq * 4) * 256 + xq * 4];
        a += f[p] * m;
    }
    s[tid] = a; __syncthreads();
    for (int st = TPB / 2; st > 0; st >>= 1) {
        if (tid < st) s[tid] += s[tid + st];
        __syncthreads();
    }
    if (tid == 0) mean[b * 512 + ch] = s[0] / fmaxf(cnt[b * 5 + region - 1], 1.0f);
}

// ---------------------------------------------------------------------------
// bar = (f - mean)*m ; n = ||bar||_2 over channels ; out[p][c] = bar/max(n,1e-12)
// ---------------------------------------------------------------------------
__global__ void k_region_norm(const float* __restrict__ feat, const float* __restrict__ mask,
                              const float* __restrict__ mean, _Float16* __restrict__ out, int region)
{
    int id = blockIdx.x * TPB + threadIdx.x;   // 0..8191
    int b = id >> 12, p = id & 4095;
    int yq = p >> 6, xq = p & 63;
    float m = mask[(((size_t)b * 6 + region) * 256 + yq * 4) * 256 + xq * 4];
    const float* f = feat + (size_t)b * 512 * 4096 + p;
    const float* mn = mean + b * 512;
    float ss = 0.f;
    for (int c = 0; c < 512; ++c) {
        float v = (f[(size_t)c * 4096] - mn[c]) * m;
        ss += v * v;
    }
    float inv = 1.0f / fmaxf(sqrtf(ss), 1e-12f);
    _Float16* o = out + ((size_t)b * 4096 + p) * 512;
    for (int c = 0; c < 512; ++c) {
        float v = (f[(size_t)c * 4096] - mn[c]) * m;
        o[c] = (_Float16)(v * inv);
    }
}

// ---------------------------------------------------------------------------
// TDM: 1-D tile load (contiguous 8192 x 2B = one 16px x 512ch fp16 K-tile)
// D# per CDNA5 ISA ch.8: group0 {count=1, lds_addr, global_addr, type=2},
// group1 {data_size=1(2B), tensor_dim0=tile_dim0=8192, stride=8192}.
// ---------------------------------------------------------------------------
#if USE_TDM
__device__ __forceinline__ void tdm_load_tile(const _Float16* gsrc, void* lds_dst)
{
    unsigned long long ga = (unsigned long long)gsrc;
    u32x4 g0;
    g0.x = 1u;                                           // count=1, user mode
    g0.y = (unsigned)(unsigned long long)lds_dst;        // LDS byte offset (low 32b of generic)
    g0.z = (unsigned)(ga & 0xFFFFFFFFull);               // global_addr[31:0]
    g0.w = (unsigned)((ga >> 32) & 0x01FFFFFFull) | (2u << 30);  // addr[56:32] | type=2
    i32x8 g1;
    g1[0] = 0x00010000;   // wg_mask=0, data_size=1 (2 bytes)
    g1[1] = 0x20000000;   // tensor_dim0[15:0]=8192 in bits[63:48]
    g1[2] = 0x00010000;   // tensor_dim0[31:16]=0, tensor_dim1[15:0]=1
    g1[3] = 0x20000000;   // tensor_dim1[31:16]=0, tile_dim0=8192
    g1[4] = 0;            // tile_dim1=0 (1-D), tile_dim2=0
    g1[5] = 0x00002000;   // tensor_dim0_stride = 8192
    g1[6] = 0;
    g1[7] = 0;
    i32x4 z4; z4[0] = 0; z4[1] = 0; z4[2] = 0; z4[3] = 0;
#if TDM_ARITY == 6
    i32x8 z8; for (int i = 0; i < 8; ++i) z8[i] = 0;
    __builtin_amdgcn_tensor_load_to_lds(g0, g1, z4, z4, z8, 0);
#else
    __builtin_amdgcn_tensor_load_to_lds(g0, g1, z4, z4, 0);
#endif
}
#endif

// ---------------------------------------------------------------------------
// Flash-attention region transfer with v_wmma_f32_16x16x32_f16.
// Q held in regs (16 x v16h per wave, 16 target rows/wave, 8 waves/block).
// K-tiles double-buffered in LDS; staged by the Tensor Data Mover (wave 0
// issues tensor_load_to_lds one tile ahead, s_wait_tensorcnt + barrier
// hand-off) with a plain-load fallback when the builtin is unavailable.
// grid (32, 2) block 256.
// ---------------------------------------------------------------------------
__global__ void __launch_bounds__(256)
k_flash_attn(const _Float16* __restrict__ TN, const _Float16* __restrict__ SN,
             const float* __restrict__ scq, const float* __restrict__ tmask,
             const float* __restrict__ smask, const float* __restrict__ cntT,
             const float* __restrict__ cntS, float* __restrict__ canvas, int region)
{
    __shared__ __align__(16) _Float16 Ks[2][16 * 512];   // 2 x 16KB
    __shared__ float Vs[48];
    __shared__ float Ms[16];

    const int b = blockIdx.y;
    const int tid = threadIdx.x;
    const int lane = tid & 31;
    const int wv = tid >> 5;
    const int col = lane & 15;
    const int hi = lane >> 4;
    const int rowbase = blockIdx.x * 128 + wv * 16;

    const bool valid = (cntT[b * 5 + region - 1] > 1.0f) && (cntS[b * 5 + region - 1] > 1.0f);

    // A-matrix fragments: lane holds row (lane&15); k-subsets per hi half.
    v16h qf[16];
    {
        const _Float16* qrow = TN + ((size_t)b * 4096 + rowbase + col) * 512;
        #pragma unroll
        for (int kc = 0; kc < 16; ++kc) {
            v8h lo = *(const v8h*)(qrow + kc * 32 + 8 * hi);
            v8h hp = *(const v8h*)(qrow + kc * 32 + 16 + 8 * hi);
            v16h a;
            #pragma unroll
            for (int i = 0; i < 8; ++i) { a[i] = lo[i]; a[i + 8] = hp[i]; }
            qf[kc] = a;
        }
    }

    float runmax[8], lsum[8], o0[8], o1[8], o2[8];
    #pragma unroll
    for (int r = 0; r < 8; ++r) { runmax[r] = -1e30f; lsum[r] = 0.f; o0[r] = 0.f; o1[r] = 0.f; o2[r] = 0.f; }

    const _Float16* SNb = SN + (size_t)b * 4096 * 512;

#if USE_TDM
    if (wv == 0) tdm_load_tile(SNb, &Ks[0][0]);   // prime the pipeline
#endif

    for (int tile = 0; tile < 256; ++tile) {
        const int cur = tile & 1;
#if USE_TDM
        if (wv == 0 && tile + 1 < 256)
            tdm_load_tile(SNb + (size_t)(tile + 1) * 16 * 512, &Ks[cur ^ 1][0]);
#else
        {   // cooperative stage: 256 thr x 64B
            int kcol = tid >> 4;
            int seg = tid & 15;
            const v8h* src = (const v8h*)(SNb + ((size_t)tile * 16 + kcol) * 512 + seg * 32);
            v8h* dst = (v8h*)(&Ks[cur][kcol * 512 + seg * 32]);
            dst[0] = src[0]; dst[1] = src[1]; dst[2] = src[2]; dst[3] = src[3];
        }
#endif
        if (tid < 16) {
            int sp = tile * 16 + tid;
            int sy = sp >> 6, sx = sp & 63;
            Ms[tid] = smask[(((size_t)b * 6 + region) * 256 + sy * 4) * 256 + sx * 4];
            Vs[tid]      = scq[((size_t)b * 3 + 0) * 4096 + sp];
            Vs[16 + tid] = scq[((size_t)b * 3 + 1) * 4096 + sp];
            Vs[32 + tid] = scq[((size_t)b * 3 + 2) * 4096 + sp];
        }
#if USE_TDM
        if (wv == 0) {
            if (tile + 1 < 256) __builtin_amdgcn_s_wait_tensorcnt(1);   // current tile landed
            else                __builtin_amdgcn_s_wait_tensorcnt(0);   // drain on last tile
        }
#endif
        __syncthreads();
        float mcol = Ms[col];
        float vc0 = Vs[col], vc1 = Vs[16 + col], vc2 = Vs[32 + col];

        v8f c = {0.f, 0.f, 0.f, 0.f, 0.f, 0.f, 0.f, 0.f};
        #pragma unroll
        for (int kc = 0; kc < 16; ++kc) {
            v8h lo = *(const v8h*)(&Ks[cur][col * 512 + kc * 32 + 8 * hi]);
            v8h hp = *(const v8h*)(&Ks[cur][col * 512 + kc * 32 + 16 + 8 * hi]);
            v16h bb;
            #pragma unroll
            for (int i = 0; i < 8; ++i) { bb[i] = lo[i]; bb[i + 8] = hp[i]; }
            c = __builtin_amdgcn_wmma_f32_16x16x32_f16(false, qf[kc], false, bb,
                                                       (short)0, c, false, false);
        }

        bool masked = !(mcol > 0.0f);
        #pragma unroll
        for (int r = 0; r < 8; ++r) {
            float sv = c[r];
            float smax = masked ? -1e30f : sv;
            smax = fmaxf(smax, __shfl_xor(smax, 1));
            smax = fmaxf(smax, __shfl_xor(smax, 2));
            smax = fmaxf(smax, __shfl_xor(smax, 4));
            smax = fmaxf(smax, __shfl_xor(smax, 8));
            float nm = fmaxf(runmax[r], smax);
            float scale = __expf(runmax[r] - nm);
            float pe = masked ? 0.0f : __expf(sv - nm);
            lsum[r] = lsum[r] * scale + pe;
            o0[r] = o0[r] * scale + pe * vc0;
            o1[r] = o1[r] * scale + pe * vc1;
            o2[r] = o2[r] * scale + pe * vc2;
            runmax[r] = nm;
        }
        __syncthreads();   // all waves done with Ks[cur] before it is refilled
    }

    #pragma unroll
    for (int r = 0; r < 8; ++r) {
        lsum[r] += __shfl_xor(lsum[r], 1); lsum[r] += __shfl_xor(lsum[r], 2);
        lsum[r] += __shfl_xor(lsum[r], 4); lsum[r] += __shfl_xor(lsum[r], 8);
        o0[r] += __shfl_xor(o0[r], 1); o0[r] += __shfl_xor(o0[r], 2);
        o0[r] += __shfl_xor(o0[r], 4); o0[r] += __shfl_xor(o0[r], 8);
        o1[r] += __shfl_xor(o1[r], 1); o1[r] += __shfl_xor(o1[r], 2);
        o1[r] += __shfl_xor(o1[r], 4); o1[r] += __shfl_xor(o1[r], 8);
        o2[r] += __shfl_xor(o2[r], 1); o2[r] += __shfl_xor(o2[r], 2);
        o2[r] += __shfl_xor(o2[r], 4); o2[r] += __shfl_xor(o2[r], 8);
    }

    if (valid) {
        #pragma unroll
        for (int r = 0; r < 8; ++r) {
            if (col == r) {   // writer lane for row r + 8*hi of this wave tile
                int t = rowbase + r + 8 * hi;
                int ty = t >> 6, tx = t & 63;
                float tm = tmask[(((size_t)b * 6 + region) * 256 + ty * 4) * 256 + tx * 4];
                if (tm > 0.0f && lsum[r] > 0.0f) {
                    float inv = 1.0f / lsum[r];
                    canvas[((size_t)b * 3 + 0) * 4096 + t] = o0[r] * inv;
                    canvas[((size_t)b * 3 + 1) * 4096 + t] = o1[r] * inv;
                    canvas[((size_t)b * 3 + 2) * 4096 + t] = o2[r] * inv;
                }
            }
        }
    }
}

// ---------------------------------------------------------------------------
// UNet input: ch0 = target_gray, ch1..3 = color_map
// ---------------------------------------------------------------------------
__global__ void k_concat(const float* __restrict__ g, const float* __restrict__ cm,
                         float* __restrict__ o)
{
    long long idx = (long long)blockIdx.x * blockDim.x + threadIdx.x;
    long long total = 2LL * 4 * 65536;
    if (idx >= total) return;
    int p = (int)(idx % 65536); long long t = idx / 65536;
    int c = (int)(t % 4); int b = (int)(t / 4);
    o[idx] = (c == 0) ? g[(size_t)b * 65536 + p]
                      : cm[((size_t)b * 3 + (c - 1)) * 65536 + p];
}

// ---------------------------------------------------------------------------
// Host side
// ---------------------------------------------------------------------------
static void run_fpn(const float* x, const float* const* P, float* C0, float* C1,
                    float* C2, float* C3, float* F, hipStream_t stream)
{
    long long n;
    n = 2LL * 32 * 256 * 256;
    k_refpad_conv7<<<GRID1(n), TPB, 0, stream>>>(x, P[0], P[1], C0, 2, 3, 32, 256, 0);
    n = 2LL * 64 * 128 * 128;
    k_conv2d<<<GRID1(n), TPB, 0, stream>>>(C0, P[2], P[3], C1, 2, 32, 256, 256, 64, 128, 128, 3, 3, 2, 1);
    k_inorm<<<dim3(2 * 64), TPB, 0, stream>>>(C1, 128 * 128, 1);
    n = 2LL * 128 * 64 * 64;
    k_conv2d<<<GRID1(n), TPB, 0, stream>>>(C1, P[4], P[5], C0, 2, 64, 128, 128, 128, 64, 64, 3, 3, 2, 1);
    k_inorm<<<dim3(2 * 128), TPB, 0, stream>>>(C0, 64 * 64, 1);
    n = 2LL * 256 * 32 * 32;
    k_conv2d<<<GRID1(n), TPB, 0, stream>>>(C0, P[6], P[7], C2, 2, 128, 64, 64, 256, 32, 32, 3, 3, 2, 1);
    k_inorm<<<dim3(2 * 256), TPB, 0, stream>>>(C2, 32 * 32, 1);                  // x3 kept in C2
    n = 2LL * 512 * 16 * 16;
    k_conv2d<<<GRID1(n), TPB, 0, stream>>>(C2, P[8], P[9], C1, 2, 256, 32, 32, 512, 16, 16, 3, 3, 2, 1);
    k_inorm<<<dim3(2 * 512), TPB, 0, stream>>>(C1, 16 * 16, 1);                  // x4
    n = 2LL * 512 * 32 * 32;
    k_tconv3<<<GRID1(n), TPB, 0, stream>>>(C1, P[10], P[11], C0, 2, 512, 16, 16, 512);
    k_inorm<<<dim3(2 * 512), TPB, 0, stream>>>(C0, 32 * 32, 1);                  // y = u4 block
    n = 2LL * 512 * 32 * 32;
    k_conv2d<<<GRID1(n), TPB, 0, stream>>>(C2, P[14], P[15], C3, 2, 256, 32, 32, 512, 32, 32, 1, 1, 1, 0);
    k_add<<<GRID1(n), TPB, 0, stream>>>(C0, C3, n);                              // y + skip
    n = 2LL * 512 * 64 * 64;
    k_tconv3<<<GRID1(n), TPB, 0, stream>>>(C0, P[12], P[13], F, 2, 512, 32, 32, 512);
    k_inorm<<<dim3(2 * 512), TPB, 0, stream>>>(F, 64 * 64, 1);                   // feature
}

extern "C" void kernel_launch(void* const* d_in, const int* in_sizes, int n_in,
                              void* d_out, int out_size, void* d_ws, size_t ws_size,
                              hipStream_t stream)
{
    (void)in_sizes; (void)n_in; (void)out_size; (void)ws_size;

    const float* src_color = (const float*)d_in[0];
    const float* src_mask  = (const float*)d_in[1];
    const float* tgt_color = (const float*)d_in[2];
    const float* tgt_gray  = (const float*)d_in[3];
    const float* tgt_mask  = (const float*)d_in[4];
    const float* P[32];
    for (int i = 0; i < 32; ++i) P[i] = (const float*)d_in[5 + i];
    // P: 0 f_in_w 1 f_in_b 2 f_d1_w 3 b 4 f_d2_w 5 b 6 f_d3_w 7 b 8 f_d4_w 9 b
    //    10 f_u4_w 11 b 12 f_u3_w 13 b 14 f_sk_w 15 b
    //    16 u_in_w 17 b 18 u_d1_w 19 b 20 u_d2_w 21 b 22 u_d3_w 23 b
    //    24 u_u3_w 25 b 26 u_u2_w 27 b 28 u_u1_w 29 b 30 u_out_w 31 b

    // Workspace layout (floats). Total ~34.13M floats = ~131 MB.
    float* ws  = (float*)d_ws;
    float* FT  = ws + 0;          // 2x512x4096
    float* FS  = ws + 4194304;    // 2x512x4096
    float* C0  = ws + 8388608;    // big scratch (8.39M el)
    float* C1  = ws + 16777216;   // big scratch (8.39M el)
    float* C2  = ws + 25165824;   // med scratch (2.10M el)
    float* C3  = ws + 27262976;   // med scratch (2.10M el)
    float* SCQ = ws + 29360128;   // 2x3x4096
    float* CNV = ws + 29384704;   // 2x3x4096
    float* UIN = ws + 29409280;   // 2x4x65536
    float* CT  = ws + 29933568;   // 10 used
    float* CS  = ws + 29933584;   // 10 used
    float* MT  = ws + 29933600;   // 2x512
    float* MS  = ws + 29934624;   // 2x512
    _Float16* TN = (_Float16*)(ws + 29935648);   // 2x4096x512 halves
    _Float16* SN = (_Float16*)(ws + 32032800);   // 2x4096x512 halves

    float* blend = (float*)d_out;              // 2x3x256x256
    float* cmap  = (float*)d_out + 393216;     // 2x3x256x256

    // ---- FPN on source and target colors ----
    run_fpn(src_color, P, C0, C1, C2, C3, FS, stream);
    run_fpn(tgt_color, P, C0, C1, C2, C3, FT, stream);

    // ---- color map ----
    k_mask_count<<<dim3(5, 2), TPB, 0, stream>>>(tgt_mask, CT);
    k_mask_count<<<dim3(5, 2), TPB, 0, stream>>>(src_mask, CS);
    k_resize_bilinear<<<GRID1(2LL * 3 * 64 * 64), TPB, 0, stream>>>(src_color, SCQ, 2, 3, 256, 256, 64, 64);
    k_zero<<<GRID1(24576LL), TPB, 0, stream>>>(CNV, 24576);

    for (int r = 1; r <= 5; ++r) {
        k_region_mean<<<dim3(512, 2), TPB, 0, stream>>>(FT, tgt_mask, CT, MT, r);
        k_region_mean<<<dim3(512, 2), TPB, 0, stream>>>(FS, src_mask, CS, MS, r);
        k_region_norm<<<dim3(32), TPB, 0, stream>>>(FT, tgt_mask, MT, TN, r);
        k_region_norm<<<dim3(32), TPB, 0, stream>>>(FS, src_mask, MS, SN, r);
        k_flash_attn<<<dim3(32, 2), TPB, 0, stream>>>(TN, SN, SCQ, tgt_mask, src_mask, CT, CS, CNV, r);
    }
    k_resize_bilinear<<<GRID1(2LL * 3 * 256 * 256), TPB, 0, stream>>>(CNV, cmap, 2, 3, 64, 64, 256, 256);

    // ---- UNet ----
    long long n;
    k_concat<<<GRID1(2LL * 4 * 65536), TPB, 0, stream>>>(tgt_gray, cmap, UIN);
    n = 2LL * 64 * 256 * 256;
    k_refpad_conv7<<<GRID1(n), TPB, 0, stream>>>(UIN, P[16], P[17], C0, 2, 4, 64, 256, 0);
    n = 2LL * 128 * 128 * 128;
    k_conv2d<<<GRID1(n), TPB, 0, stream>>>(C0, P[18], P[19], C1, 2, 64, 256, 256, 128, 128, 128, 3, 3, 2, 1);
    k_inorm<<<dim3(2 * 128), TPB, 0, stream>>>(C1, 128 * 128, 1);
    n = 2LL * 256 * 64 * 64;
    k_conv2d<<<GRID1(n), TPB, 0, stream>>>(C1, P[20], P[21], C2, 2, 128, 128, 128, 256, 64, 64, 3, 3, 2, 1);
    k_inorm<<<dim3(2 * 256), TPB, 0, stream>>>(C2, 64 * 64, 1);
    n = 2LL * 512 * 32 * 32;
    k_conv2d<<<GRID1(n), TPB, 0, stream>>>(C2, P[22], P[23], C3, 2, 256, 64, 64, 512, 32, 32, 3, 3, 2, 1);
    k_inorm<<<dim3(2 * 512), TPB, 0, stream>>>(C3, 32 * 32, 1);
    n = 2LL * 256 * 64 * 64;
    k_tconv3<<<GRID1(n), TPB, 0, stream>>>(C3, P[24], P[25], C0, 2, 512, 32, 32, 256);
    k_inorm<<<dim3(2 * 256), TPB, 0, stream>>>(C0, 64 * 64, 1);
    n = 2LL * 128 * 128 * 128;
    k_tconv3<<<GRID1(n), TPB, 0, stream>>>(C0, P[26], P[27], C1, 2, 256, 64, 64, 128);
    k_inorm<<<dim3(2 * 128), TPB, 0, stream>>>(C1, 128 * 128, 1);
    n = 2LL * 64 * 256 * 256;
    k_tconv3<<<GRID1(n), TPB, 0, stream>>>(C1, P[28], P[29], C0, 2, 128, 128, 128, 64);
    k_inorm<<<dim3(2 * 64), TPB, 0, stream>>>(C0, 256 * 256, 1);
    n = 2LL * 3 * 256 * 256;
    k_refpad_conv7<<<GRID1(n), TPB, 0, stream>>>(C0, P[30], P[31], blend, 2, 64, 3, 256, 1);
}
